// STCrossTransformer_22204980921036
// MI455X (gfx1250) — matrix operand order repaired and checked by reading, not compiled, not run on hardware
//
#include <hip/hip_runtime.h>
#include <hip/hip_bf16.h>

typedef __bf16 bf16_t;
typedef __attribute__((ext_vector_type(16))) __bf16 v16bf;
typedef __attribute__((ext_vector_type(8)))  __bf16 v8bf;
typedef __attribute__((ext_vector_type(2)))  __bf16 v2bf;
typedef __attribute__((ext_vector_type(8)))  float  v8f;

#define DIM 768
#define NH  12
#define HD  64
#define LQ  1568           // 196 * 8
#define NB  4
#define M_ROWS (NB*LQ)     // 6272
#define MT32 (M_ROWS/32)   // 196 (32-row GEMM tiles)
#define QT32 (LQ/32)       // 49  (32-query attention tiles)

// ---------- helpers ----------

// A-operand (16x32 bf16) per-lane fragment: lanes 0-15 hold K {0..15},
// lanes 16-31 hold K {16..31}; pass c0 = 8*(lane>=16).
static __device__ inline v16bf ld_a_chunks(const bf16_t* p, int c0) {
  v8bf lo = *(const v8bf*)(p + 2 * c0);
  v8bf hi = *(const v8bf*)(p + 2 * c0 + 8);
  return __builtin_shufflevector(lo, hi, 0,1,2,3,4,5,6,7,8,9,10,11,12,13,14,15);
}

static __device__ inline v8f wmma_bf16(v16bf a, v16bf b, v8f c) {
  return __builtin_amdgcn_wmma_f32_16x16x32_bf16(false, a, false, b, (short)0, c, false, false);
}

static __device__ inline int pack2(float a, float b) {
  v2bf t; t[0] = (bf16_t)a; t[1] = (bf16_t)b;
  return __builtin_bit_cast(int, t);
}

// ---------- prep kernels ----------

// seq[b][n*8+tt][d] = xpat[n][b*8+tt][d] + spos[n][d] + tpos[tt][d]   (fp32 -> bf16)
__global__ void build_seq_kernel(const float* __restrict__ xpat,
                                 const float* __restrict__ spos,
                                 const float* __restrict__ tpos,
                                 bf16_t* __restrict__ seq) {
  size_t idx = (size_t)blockIdx.x * blockDim.x + threadIdx.x;
  const size_t total = (size_t)NB * LQ * (DIM / 4);
  if (idx >= total) return;
  int d4 = (int)(idx % (DIM / 4));
  size_t rest = idx / (DIM / 4);
  int l = (int)(rest % LQ);
  int b = (int)(rest / LQ);
  int n = l >> 3, tt = l & 7;
  int d = d4 * 4;
  const float4 xv = *(const float4*)(xpat + ((size_t)n * (NB * 8) + b * 8 + tt) * DIM + d);
  const float4 sv = *(const float4*)(spos + (size_t)n * DIM + d);
  const float4 tv = *(const float4*)(tpos + (size_t)tt * DIM + d);
  bf16_t* o = seq + ((size_t)b * LQ + l) * DIM + d;
  o[0] = (bf16_t)(xv.x + sv.x + tv.x);
  o[1] = (bf16_t)(xv.y + sv.y + tv.y);
  o[2] = (bf16_t)(xv.z + sv.z + tv.z);
  o[3] = (bf16_t)(xv.w + sv.w + tv.w);
}

__global__ void f2bf_kernel(const float* __restrict__ x, bf16_t* __restrict__ y, int n) {
  int i = blockIdx.x * blockDim.x + threadIdx.x;
  if (i < n) y[i] = (bf16_t)x[i];
}

__global__ void copy_cls_kernel(const float* __restrict__ t_x, float* __restrict__ out) {
  int i = blockIdx.x * blockDim.x + threadIdx.x;
  if (i < 32 * DIM) out[i] = t_x[i];
}

// ---------- GEMM: C[M x N] = X[M x 768](bf16) * W[N x 768]^T + bias ----------
// one wave -> 32x64 output tile (8 WMMA accumulators, A reused x4, B reused x2).
// mode 0: proj -> d_out fp32 with output permute
// mode 1: Q -> q_bf [bh][l][64], scaled by 1/8
// mode 2: KV -> k_bf [bh][l][64], vT_bf [bh][d][l]
__global__ __launch_bounds__(128)
void gemm_kernel(const bf16_t* __restrict__ X, const bf16_t* __restrict__ W,
                 const float* __restrict__ bias, int nt64, int mode,
                 bf16_t* __restrict__ out_bf0, bf16_t* __restrict__ out_bf1,
                 float* __restrict__ out_f) {
  const int lane = threadIdx.x & 31;
  const int w = blockIdx.x * (blockDim.x >> 5) + (threadIdx.x >> 5);
  const int mtile = w / nt64;
  const int ntile = w % nt64;
  if (mtile >= MT32) return;
  const int l15 = lane & 15;
  const int hi  = lane >> 4;
  const int c0  = hi * 8;

  const bf16_t* xrow0 = X + (size_t)(mtile * 32 + l15) * DIM;
  const bf16_t* xrow1 = xrow0 + (size_t)16 * DIM;
  const bf16_t* wbase = W + (size_t)(ntile * 64 + l15) * DIM;

  v8f acc[2][4] = {};
#pragma unroll 2
  for (int k = 0; k < DIM; k += 32) {
    v16bf a0 = ld_a_chunks(xrow0 + k, c0);
    v16bf a1 = ld_a_chunks(xrow1 + k, c0);
#pragma unroll
    for (int j = 0; j < 4; j++) {
      v16bf bv = *(const v16bf*)(wbase + (size_t)j * 16 * DIM + k + 16 * hi);
      acc[0][j] = wmma_bf16(a0, bv, acc[0][j]);
      acc[1][j] = wmma_bf16(a1, bv, acc[1][j]);
    }
  }

#pragma unroll
  for (int mt = 0; mt < 2; mt++) {
#pragma unroll
    for (int j = 0; j < 4; j++) {
      int n = ntile * 64 + j * 16 + l15;
      float bv = bias[n];
#pragma unroll
      for (int i = 0; i < 8; i++) {
        int r = mtile * 32 + mt * 16 + i + 8 * hi;  // row = b*1568 + l
        float val = acc[mt][j][i] + bv;
        int b = r / LQ, l = r % LQ;
        if (mode == 0) {
          int nidx = l >> 3, tt = l & 7;            // out[(1+nidx)][b*8+tt][n]
          out_f[(((size_t)(1 + nidx)) * 32 + b * 8 + tt) * DIM + n] = val;
        } else if (mode == 1) {
          int h = n >> 6, dh = n & 63;
          out_bf0[(((size_t)(b * NH + h)) * LQ + l) * HD + dh] = (bf16_t)(val * 0.125f);
        } else {
          if (n < DIM) {                             // K part
            int h = n >> 6, dh = n & 63;
            out_bf0[(((size_t)(b * NH + h)) * LQ + l) * HD + dh] = (bf16_t)val;
          } else {                                   // V part, stored transposed
            int c = n - DIM; int h = c >> 6, dh = c & 63;
            out_bf1[(((size_t)(b * NH + h)) * HD + dh) * LQ + l] = (bf16_t)val;
          }
        }
      }
    }
  }
}

// ---------- flash attention: one wave = 32 queries of one (b,h) ----------
__global__ __launch_bounds__(128)
void attn_kernel(const bf16_t* __restrict__ Q,   // [48][1568][64] pre-scaled
                 const bf16_t* __restrict__ K,   // [48][1568][64]
                 const bf16_t* __restrict__ Vt,  // [48][64][1568]
                 bf16_t* __restrict__ O) {       // [4][1568][768]
  const int lane = threadIdx.x & 31;
  const int w = blockIdx.x * (blockDim.x >> 5) + (threadIdx.x >> 5);
  const int bh = w / QT32;
  const int qt = w % QT32;
  if (bh >= NB * NH) return;
  const int l15 = lane & 15, hi = lane >> 4, c0 = hi * 8;

  // Q as B-operand of S^T = K x Q^T : column = query, elements = head dims
  v16bf qb[2][2];
#pragma unroll
  for (int qi = 0; qi < 2; qi++) {
    const bf16_t* qp = Q + ((size_t)bh * LQ + qt * 32 + qi * 16 + l15) * HD;
    qb[qi][0] = *(const v16bf*)(qp + 16 * hi);        // d 0..31
    qb[qi][1] = *(const v16bf*)(qp + 32 + 16 * hi);   // d 32..63
  }

  const bf16_t* Kb = K  + (size_t)bh * LQ * HD;
  const bf16_t* Vb = Vt + (size_t)bh * HD * LQ;

  v8f o[2][4] = {};                                    // [qtile][dimtile]
  float mrun[2] = {-3.0e38f, -3.0e38f}, lrun[2] = {0.0f, 0.0f};

  for (int k0 = 0; k0 < LQ; k0 += 32) {
    const bf16_t* kr0 = Kb + (size_t)(k0 + l15) * HD;
    const bf16_t* kr1 = Kb + (size_t)(k0 + 16 + l15) * HD;
    v16bf ka00 = ld_a_chunks(kr0,      c0);
    v16bf ka01 = ld_a_chunks(kr0 + 32, c0);
    v16bf ka10 = ld_a_chunks(kr1,      c0);
    v16bf ka11 = ld_a_chunks(kr1 + 32, c0);

    v8f s[2][2] = {};                                  // [keytile][qtile]
#pragma unroll
    for (int qi = 0; qi < 2; qi++) {
      s[0][qi] = wmma_bf16(ka00, qb[qi][0], s[0][qi]);
      s[0][qi] = wmma_bf16(ka01, qb[qi][1], s[0][qi]);
      s[1][qi] = wmma_bf16(ka10, qb[qi][0], s[1][qi]);
      s[1][qi] = wmma_bf16(ka11, qb[qi][1], s[1][qi]);
    }

    union { v16bf v; int d[8]; } pb[2];
#pragma unroll
    for (int qi = 0; qi < 2; qi++) {
      // online softmax per query (column): 16 key rows split across lane-halves
      float tm = s[0][qi][0];
#pragma unroll
      for (int i = 1; i < 8; i++) tm = fmaxf(tm, s[0][qi][i]);
#pragma unroll
      for (int i = 0; i < 8; i++) tm = fmaxf(tm, s[1][qi][i]);
      tm = fmaxf(tm, __shfl_xor(tm, 16, 32));
      float mnew  = fmaxf(mrun[qi], tm);
      float alpha = __expf(mrun[qi] - mnew);
      float p0[8], p1[8], ps = 0.0f;
#pragma unroll
      for (int i = 0; i < 8; i++) {
        p0[i] = __expf(s[0][qi][i] - mnew);
        p1[i] = __expf(s[1][qi][i] - mnew);
        ps += p0[i] + p1[i];
      }
      ps += __shfl_xor(ps, 16, 32);
      lrun[qi] = lrun[qi] * alpha + ps;
      mrun[qi] = mnew;
#pragma unroll
      for (int j = 0; j < 4; j++)
#pragma unroll
        for (int i = 0; i < 8; i++) o[qi][j][i] *= alpha;

      // P^T as B-operand (32 keys x 16 queries): lane-lo needs keys 0..15,
      // lane-hi keys 16..31 -> one xor-16 swap per packed dword pair.
      int u0[4], u1[4], x0[4], x1[4];
#pragma unroll
      for (int i = 0; i < 4; i++) {
        u0[i] = pack2(p0[2 * i], p0[2 * i + 1]);
        u1[i] = pack2(p1[2 * i], p1[2 * i + 1]);
      }
#pragma unroll
      for (int i = 0; i < 4; i++) {
        x0[i] = __shfl_xor(u0[i], 16, 32);
        x1[i] = __shfl_xor(u1[i], 16, 32);
      }
#pragma unroll
      for (int i = 0; i < 4; i++) {
        pb[qi].d[i]     = hi ? x1[i] : u0[i];
        pb[qi].d[4 + i] = hi ? u1[i] : x0[i];
      }
    }

    // O^T += V^T x P^T ; V fragment reused across both query tiles
#pragma unroll
    for (int j = 0; j < 4; j++) {
      const bf16_t* vr = Vb + (size_t)(j * 16 + l15) * LQ + k0;
      v16bf va = ld_a_chunks(vr, c0);
      o[0][j] = wmma_bf16(va, pb[0].v, o[0][j]);
      o[1][j] = wmma_bf16(va, pb[1].v, o[1][j]);
    }
  }

  int b = bh / NH, h = bh % NH;
#pragma unroll
  for (int qi = 0; qi < 2; qi++) {
    float inv = 1.0f / lrun[qi];
    int q = qt * 32 + qi * 16 + l15;
    bf16_t* orow = O + ((size_t)b * LQ + q) * DIM + h * HD + 8 * hi;
#pragma unroll
    for (int j = 0; j < 4; j++) {
      v8bf st;
#pragma unroll
      for (int i = 0; i < 8; i++) st[i] = (bf16_t)(o[qi][j][i] * inv);
      *(v8bf*)(orow + j * 16) = st;
    }
  }
}

// ---------- launcher ----------
extern "C" void kernel_launch(void* const* d_in, const int* in_sizes, int n_in,
                              void* d_out, int out_size, void* d_ws, size_t ws_size,
                              hipStream_t stream) {
  (void)in_sizes; (void)n_in; (void)out_size; (void)ws_size;
  const float* s_x        = (const float*)d_in[0];
  const float* t_x        = (const float*)d_in[1];
  const float* clip_space = (const float*)d_in[2];
  const float* vmae_space = (const float*)d_in[3];
  const float* clip_temp  = (const float*)d_in[4];
  const float* vmae_temp  = (const float*)d_in[5];
  const float* Wq         = (const float*)d_in[6];
  const float* q_bias     = (const float*)d_in[7];
  const float* Wkv        = (const float*)d_in[8];
  const float* kv_bias    = (const float*)d_in[9];
  const float* Wproj      = (const float*)d_in[10];
  const float* proj_bias  = (const float*)d_in[11];
  float* out = (float*)d_out;

  char* ws = (char*)d_ws;
  size_t off = 0;
  auto take = [&](size_t bytes) -> char* {
    char* p = ws + off;
    off = (off + bytes + 255) & ~(size_t)255;
    return p;
  };
  const size_t SEQ_E = (size_t)NB * LQ * DIM;
  bf16_t* t_seq  = (bf16_t*)take(SEQ_E * 2);
  bf16_t* s_seq  = (bf16_t*)take(SEQ_E * 2);
  bf16_t* wq_bf  = (bf16_t*)take((size_t)DIM * DIM * 2);
  bf16_t* wkv_bf = (bf16_t*)take((size_t)2 * DIM * DIM * 2);
  bf16_t* wp_bf  = (bf16_t*)take((size_t)DIM * DIM * 2);
  bf16_t* q_bf   = (bf16_t*)take(SEQ_E * 2);
  bf16_t* k_bf   = (bf16_t*)take(SEQ_E * 2);
  bf16_t* vt_bf  = (bf16_t*)take(SEQ_E * 2);
  bf16_t* ao_bf  = (bf16_t*)take(SEQ_E * 2);

  int bs_blocks = (int)((SEQ_E / 4 + 255) / 256);
  build_seq_kernel<<<bs_blocks, 256, 0, stream>>>(t_x + (size_t)1 * 32 * DIM, vmae_space, vmae_temp, t_seq);
  build_seq_kernel<<<bs_blocks, 256, 0, stream>>>(s_x + (size_t)2 * 32 * DIM, clip_space, clip_temp, s_seq);
  f2bf_kernel<<<(DIM * DIM + 255) / 256, 256, 0, stream>>>(Wq, wq_bf, DIM * DIM);
  f2bf_kernel<<<(2 * DIM * DIM + 255) / 256, 256, 0, stream>>>(Wkv, wkv_bf, 2 * DIM * DIM);
  f2bf_kernel<<<(DIM * DIM + 255) / 256, 256, 0, stream>>>(Wproj, wp_bf, DIM * DIM);
  copy_cls_kernel<<<(32 * DIM + 255) / 256, 256, 0, stream>>>(t_x, out);

  {   // Q = t_seq @ Wq^T + b, *SCALE, head-split
    int nt64 = DIM / 64, waves = MT32 * nt64;                    // 2352
    gemm_kernel<<<waves / 4, 128, 0, stream>>>(t_seq, wq_bf, q_bias, nt64, 1, q_bf, nullptr, nullptr);
  }
  {   // KV = s_seq @ Wkv^T + b, K row-major / V transposed
    int nt64 = 2 * DIM / 64, waves = MT32 * nt64;                // 4704
    gemm_kernel<<<waves / 4, 128, 0, stream>>>(s_seq, wkv_bf, kv_bias, nt64, 2, k_bf, vt_bf, nullptr);
  }
  {   // attention
    int waves = NB * NH * QT32;                                  // 2352
    attn_kernel<<<waves / 4, 128, 0, stream>>>(q_bf, k_bf, vt_bf, ao_bf);
  }
  {   // proj -> d_out (fused bias + output permutation)
    int nt64 = DIM / 64, waves = MT32 * nt64;
    gemm_kernel<<<waves / 4, 128, 0, stream>>>(ao_bf, wp_bf, proj_bias, nt64, 0, nullptr, nullptr, out);
  }
}
